// GOSE_12240656794053
// MI455X (gfx1250) — matrix operand order repaired and checked by reading, not compiled, not run on hardware
//
#include <hip/hip_runtime.h>

// ---------------------------------------------------------------------------
// MI455X (gfx1250, wave32) windowed + global attention.
// All GEMMs use v_wmma_f32_16x16x32_bf16 (fp32 accumulate).
// GEMM tiles are double-buffered: GLOBAL_LOAD_ASYNC_TO_LDS_B128 fills the
// next K-step's A tile while WMMAs consume the current one (ASYNCcnt path).
// ---------------------------------------------------------------------------

typedef __attribute__((ext_vector_type(16))) __bf16 v16bf;
typedef __attribute__((ext_vector_type(8)))  float  v8f;
typedef __attribute__((ext_vector_type(4)))  int    v4i_t;

#define BQ   8
#define CC   768
#define NHD  12
#define HDD  64
#define HWW  4096
#define NN   4097            // NC + H*W
#define MTOT (BQ * NN)       // 32776

#if defined(__has_builtin)
#if __has_builtin(__builtin_amdgcn_global_load_async_to_lds_b128)
#define USE_ASYNC_LDS 1
typedef __attribute__((address_space(1))) v4i_t* gptr_b128;
typedef __attribute__((address_space(3))) v4i_t* lptr_b128;
#endif
#if __has_builtin(__builtin_amdgcn_s_wait_asynccnt)
#define HAVE_WAIT_ASYNC_BUILTIN 1
#endif
#endif

static __device__ __forceinline__ void wait_async0() {
#if defined(USE_ASYNC_LDS)
#if defined(HAVE_WAIT_ASYNC_BUILTIN)
    __builtin_amdgcn_s_wait_asynccnt(0);
#else
    asm volatile("s_wait_asynccnt 0x0" ::: "memory");
#endif
#endif
}

static __device__ __forceinline__ v8f zero8() {
    v8f z = {0.f, 0.f, 0.f, 0.f, 0.f, 0.f, 0.f, 0.f};
    return z;
}

// ---------------------------------------------------------------------------
// f32 -> bf16 conversion
// ---------------------------------------------------------------------------
__global__ void k_f32_to_bf16(const float* __restrict__ in, __bf16* __restrict__ out, int n) {
    int i = blockIdx.x * blockDim.x + threadIdx.x;
    if (i < n) out[i] = (__bf16)in[i];
}

// ---------------------------------------------------------------------------
// Generic bf16 GEMM: C[M,Nn] = A[M,K] @ B[K,Nn]  (row-major, f32 accumulate)
// Block tile 128(M) x 64(N), 256 threads = 8 waves, each wave 32x32 (2x2 WMMA).
// Double-buffered LDS: async loads for step k+1 overlap WMMAs of step k.
// B tile stored transposed so all fragment reads are contiguous ds_load_b128.
// ---------------------------------------------------------------------------
template <bool OUT_BF16, bool HAS_BIAS>
__global__ void k_gemm_bf16(const __bf16* __restrict__ A, const __bf16* __restrict__ B,
                            float* __restrict__ Cf, __bf16* __restrict__ Cb,
                            const float* __restrict__ bias, int M, int Nn, int K) {
    __shared__ __bf16 As[2][128][40];  // row stride 80B (16B aligned)
    __shared__ __bf16 Bs[2][64][40];   // transposed: Bs[buf][n][k]

    const int tid  = threadIdx.x;
    const int lane = tid & 31;
    const int wave = tid >> 5;          // 0..7
    const int hv   = (lane >> 4) & 1;   // lane half
    const int l16  = lane & 15;
    const int wm   = (wave >> 1) * 32;  // wave M offset within tile
    const int wn   = (wave & 1) * 32;   // wave N offset within tile
    const int m0   = blockIdx.x * 128;
    const int n0   = blockIdx.y * 64;

    // per-thread staging coordinates (constant across K steps)
    const int saR0 = tid >> 2,         saK0 = (tid & 3) * 8;          // A chunk 0
    const int saR1 = (tid + 256) >> 2, saK1 = ((tid + 256) & 3) * 8;  // A chunk 1
    const int sbR  = tid >> 3,         sbN  = (tid & 7) * 8;          // B chunk

    auto stage = [&](int buf, int k0) {
        // A tile (128x32): two 16B chunks per thread, async global->LDS
        {
            int gm = m0 + saR0;
            if (gm < M) {
#if defined(USE_ASYNC_LDS)
                __builtin_amdgcn_global_load_async_to_lds_b128(
                    (gptr_b128)(A + (size_t)gm * K + k0 + saK0),
                    (lptr_b128)(&As[buf][saR0][saK0]), 0, 0);
#else
                const __bf16* src = A + (size_t)gm * K + k0 + saK0;
                #pragma unroll
                for (int e = 0; e < 8; ++e) As[buf][saR0][saK0 + e] = src[e];
#endif
            } else {
                #pragma unroll
                for (int e = 0; e < 8; ++e) As[buf][saR0][saK0 + e] = (__bf16)0.f;
            }
        }
        {
            int gm = m0 + saR1;
            if (gm < M) {
#if defined(USE_ASYNC_LDS)
                __builtin_amdgcn_global_load_async_to_lds_b128(
                    (gptr_b128)(A + (size_t)gm * K + k0 + saK1),
                    (lptr_b128)(&As[buf][saR1][saK1]), 0, 0);
#else
                const __bf16* src = A + (size_t)gm * K + k0 + saK1;
                #pragma unroll
                for (int e = 0; e < 8; ++e) As[buf][saR1][saK1 + e] = src[e];
#endif
            } else {
                #pragma unroll
                for (int e = 0; e < 8; ++e) As[buf][saR1][saK1 + e] = (__bf16)0.f;
            }
        }
        // B tile (32x64) transposed: coalesced global read, strided LDS stores
        {
            const __bf16* src = B + (size_t)(k0 + sbR) * Nn + n0 + sbN;
            #pragma unroll
            for (int e = 0; e < 8; ++e) Bs[buf][sbN + e][sbR] = src[e];
        }
    };

    v8f acc[2][2];
    acc[0][0] = zero8(); acc[0][1] = zero8();
    acc[1][0] = zero8(); acc[1][1] = zero8();

    stage(0, 0);
    int buf = 0;
    for (int k0 = 0; k0 < K; k0 += 32) {
        wait_async0();        // own async loads for `buf` complete
        __syncthreads();      // => everyone's loads/stores for `buf` complete
        if (k0 + 32 < K) stage(buf ^ 1, k0 + 32);  // overlap next tile fill

        // ---- fragments (contiguous k = 16*half + e for A and B); all reads contiguous 16B
        v16bf af[2], bfr[2];
        #pragma unroll
        for (int i = 0; i < 2; ++i) {
            int row = wm + 16 * i + l16;
            #pragma unroll
            for (int e = 0; e < 16; ++e) {
                int kl = e + 16 * hv;
                af[i][e] = As[buf][row][kl];
            }
        }
        #pragma unroll
        for (int j = 0; j < 2; ++j) {
            int col = wn + 16 * j + l16;
            #pragma unroll
            for (int e = 0; e < 16; ++e) bfr[j][e] = Bs[buf][col][e + 16 * hv];
        }
        #pragma unroll
        for (int i = 0; i < 2; ++i)
            #pragma unroll
            for (int j = 0; j < 2; ++j)
                acc[i][j] = __builtin_amdgcn_wmma_f32_16x16x32_bf16(
                    false, af[i], false, bfr[j], (short)0, acc[i][j], false, false);

        __syncthreads();      // all waves done reading `buf` before it's restaged
        buf ^= 1;
    }

    // ---- epilogue (C/D layout: M = r + 8*half, N = lane%16)
    #pragma unroll
    for (int i = 0; i < 2; ++i) {
        #pragma unroll
        for (int j = 0; j < 2; ++j) {
            #pragma unroll
            for (int r = 0; r < 8; ++r) {
                int gm = m0 + wm + 16 * i + 8 * hv + r;
                int gn = n0 + wn + 16 * j + l16;
                if (gm < M) {
                    float v = acc[i][j][r];
                    if (HAS_BIAS) v += bias[gn];
                    if (OUT_BF16) Cb[(size_t)gm * Nn + gn] = (__bf16)v;
                    else          Cf[(size_t)gm * Nn + gn] = v;
                }
            }
        }
    }
}

// ---------------------------------------------------------------------------
// Window attention: one block per (window, head, batch). 128 threads = 4 waves.
// QK^T (WMMA) -> branchless mask + softmax (LDS) -> P@V (WMMA) -> bf16 into y.
// V stored hd-major (vsT) so its B-operand fragment reads are contiguous.
// ---------------------------------------------------------------------------
__global__ void k_window_attn(const __bf16* __restrict__ qkv, const float* __restrict__ mask,
                              __bf16* __restrict__ y) {
    const int w   = blockIdx.x;   // window 0..63
    const int h   = blockIdx.y;   // head
    const int b   = blockIdx.z;   // batch
    const int tid = threadIdx.x;
    const int lane = tid & 31;
    const int wid  = tid >> 5;    // 0..3 -> 16 attention rows each
    const int hv   = (lane >> 4) & 1;
    const int l16  = lane & 15;

    __shared__ __bf16 qs[64][64];   // [token][hd]
    __shared__ __bf16 ks[64][64];   // [token][hd]
    __shared__ __bf16 vsT[64][64];  // [hd][token]  (transposed)
    __shared__ float  sc[64][65];
    __shared__ __bf16 pb[64][66];
    __shared__ float  tokf[64];

    const int wy = w >> 3, wx = w & 7;

    // ---- stage q/k/v tiles
    for (int it = 0; it < 32; ++it) {
        int idx = it * 128 + tid;
        int t = idx >> 6, d = idx & 63;
        int ty = t >> 3, tx = t & 7;
        int p  = (wy * 8 + ty) * 64 + (wx * 8 + tx);
        size_t row = (size_t)(b * NN + 1 + p) * (3 * CC);
        qs[t][d]  = qkv[row + h * HDD + d];
        ks[t][d]  = qkv[row + CC + h * HDD + d];
        vsT[d][t] = qkv[row + 2 * CC + h * HDD + d];
    }
    if (tid < 64) {
        int t = tid, ty = t >> 3, tx = t & 7;
        int p = (wy * 8 + ty) * 64 + (wx * 8 + tx);
        tokf[t] = mask[((size_t)b * HWW + p) * CC];  // per-pixel flag
    }
    __syncthreads();

    // ---- hoist mask flags into registers (branchless select later)
    float mq[8];
    #pragma unroll
    for (int r = 0; r < 8; ++r) mq[r] = tokf[wid * 16 + 8 * hv + r];

    // ---- scores = Q @ K^T * scale, masked
    v16bf aq[2];
    {
        int row = wid * 16 + l16;
        #pragma unroll
        for (int c = 0; c < 2; ++c)
            #pragma unroll
            for (int e = 0; e < 16; ++e) {
                int kl = c * 32 + e + 16 * hv;
                aq[c][e] = qs[row][kl];
            }
    }
    #pragma unroll
    for (int nt = 0; nt < 4; ++nt) {
        float mk = tokf[nt * 16 + l16];
        v8f acc = zero8();
        #pragma unroll
        for (int c = 0; c < 2; ++c) {
            v16bf bk;
            int col = nt * 16 + l16;  // key token
            #pragma unroll
            for (int e = 0; e < 16; ++e) bk[e] = ks[col][c * 32 + e + 16 * hv];
            acc = __builtin_amdgcn_wmma_f32_16x16x32_bf16(
                false, aq[c], false, bk, (short)0, acc, false, false);
        }
        #pragma unroll
        for (int r = 0; r < 8; ++r) {
            int tq = wid * 16 + 8 * hv + r;
            int tk = nt * 16 + l16;
            float v = acc[r] * 0.125f;                 // HD^-0.5
            v = (mq[r] * mk == 0.f) ? -10000.f : v;    // branchless mask
            sc[tq][tk] = v;
        }
    }
    __syncthreads();

    // ---- row softmax -> bf16 probs
    if (tid < 64) {
        float mx = -1e30f;
        for (int j = 0; j < 64; ++j) mx = fmaxf(mx, sc[tid][j]);
        float s = 0.f;
        for (int j = 0; j < 64; ++j) { float e = __expf(sc[tid][j] - mx); sc[tid][j] = e; s += e; }
        float inv = 1.f / s;
        for (int j = 0; j < 64; ++j) pb[tid][j] = (__bf16)(sc[tid][j] * inv);
    }
    __syncthreads();

    // ---- out = P @ V
    v16bf ap[2];
    {
        int row = wid * 16 + l16;
        #pragma unroll
        for (int c = 0; c < 2; ++c)
            #pragma unroll
            for (int e = 0; e < 16; ++e) {
                int kl = c * 32 + e + 16 * hv;
                ap[c][e] = pb[row][kl];
            }
    }
    #pragma unroll
    for (int nt = 0; nt < 4; ++nt) {
        v8f acc = zero8();
        #pragma unroll
        for (int c = 0; c < 2; ++c) {
            v16bf bv;
            int col = nt * 16 + l16;  // hd dim
            #pragma unroll
            for (int e = 0; e < 16; ++e) bv[e] = vsT[col][c * 32 + e + 16 * hv];
            acc = __builtin_amdgcn_wmma_f32_16x16x32_bf16(
                false, ap[c], false, bv, (short)0, acc, false, false);
        }
        #pragma unroll
        for (int r = 0; r < 8; ++r) {
            int tq = wid * 16 + 8 * hv + r;
            int d  = nt * 16 + l16;
            int ty = tq >> 3, tx = tq & 7;
            int p  = (wy * 8 + ty) * 64 + (wx * 8 + tx);
            y[(size_t)(b * NN + 1 + p) * CC + h * HDD + d] = (__bf16)acc[r];
        }
    }
}

// ---------------------------------------------------------------------------
// Global (cls) attention: NC=1 query per (b,h) over 4096 image keys.
// ---------------------------------------------------------------------------
__global__ void k_cls_attn(const __bf16* __restrict__ qkv, const float* __restrict__ mask,
                           const float* __restrict__ gmask, float* __restrict__ x_cls) {
    const int h = blockIdx.x, b = blockIdx.y;
    const int tid = threadIdx.x;  // 256

    __shared__ float qv[64];
    __shared__ float scb[HWW];
    __shared__ float red[256];
    __shared__ float part[4][64];

    if (tid < 64) qv[tid] = (float)qkv[(size_t)(b * NN) * (3 * CC) + h * HDD + tid];
    __syncthreads();

    float lmax = -1e30f;
    for (int key = tid; key < HWW; key += 256) {
        const __bf16* kr = qkv + (size_t)(b * NN + 1 + key) * (3 * CC) + CC + h * HDD;
        float s = 0.f;
        for (int d = 0; d < 64; ++d) s += qv[d] * (float)kr[d];
        s *= 0.125f;
        float tk = mask[((size_t)b * HWW + key) * CC];
        float gm = gmask[((size_t)(b * NHD + h)) * HWW + key];
        s = (tk * gm == 0.f) ? -10000.f : s;
        scb[key] = s;
        lmax = fmaxf(lmax, s);
    }
    red[tid] = lmax; __syncthreads();
    for (int off = 128; off > 0; off >>= 1) {
        if (tid < off) red[tid] = fmaxf(red[tid], red[tid + off]);
        __syncthreads();
    }
    float mx = red[0];
    __syncthreads();
    float lsum = 0.f;
    for (int key = tid; key < HWW; key += 256) {
        float e = __expf(scb[key] - mx); scb[key] = e; lsum += e;
    }
    red[tid] = lsum; __syncthreads();
    for (int off = 128; off > 0; off >>= 1) {
        if (tid < off) red[tid] += red[tid + off];
        __syncthreads();
    }
    float inv = 1.f / red[0];
    __syncthreads();

    int d = tid & 63, g = tid >> 6;  // 4 key groups x 64 dims
    float acc = 0.f;
    for (int key = g; key < HWW; key += 4)
        acc += scb[key] * (float)qkv[(size_t)(b * NN + 1 + key) * (3 * CC) + 2 * CC + h * HDD + d];
    part[g][d] = acc; __syncthreads();
    if (tid < 64)
        x_cls[(size_t)b * CC + h * HDD + tid] =
            (part[0][tid] + part[1][tid] + part[2][tid] + part[3][tid]) * inv;
}

// ---------------------------------------------------------------------------
// kv_g = x_cls @ kv_global_w  (8x768 @ 768x1536 — negligible, plain f32)
// ---------------------------------------------------------------------------
__global__ void k_kvg(const float* __restrict__ x_cls, const float* __restrict__ w,
                      float* __restrict__ kvg) {
    int i = blockIdx.x * blockDim.x + threadIdx.x;
    if (i >= BQ * 2 * CC) return;
    int b = i / (2 * CC), j = i % (2 * CC);
    float s = 0.f;
    for (int k = 0; k < CC; ++k) s += x_cls[b * CC + k] * w[k * (2 * CC) + j];
    kvg[i] = s;
}

// ---------------------------------------------------------------------------
// Build y (bf16): row 0 of each batch = x_cls; image rows += broadcast v_cls.
// (NC=1 => branch-attention softmax is identically 1 => xb == v_cls.)
// ---------------------------------------------------------------------------
__global__ void k_build_y(__bf16* __restrict__ y, const float* __restrict__ x_cls,
                          const float* __restrict__ kvg) {
    size_t i = (size_t)blockIdx.x * blockDim.x + threadIdx.x;
    if (i >= (size_t)MTOT * CC) return;
    int c = (int)(i % CC);
    size_t row = i / CC;
    int b = (int)(row / NN);
    int r = (int)(row % NN);
    if (r == 0) y[i] = (__bf16)x_cls[(size_t)b * CC + c];
    else        y[i] = (__bf16)((float)y[i] + kvg[b * (2 * CC) + CC + c]);
}

// ---------------------------------------------------------------------------
extern "C" void kernel_launch(void* const* d_in, const int* in_sizes, int n_in,
                              void* d_out, int out_size, void* d_ws, size_t ws_size,
                              hipStream_t stream) {
    (void)in_sizes; (void)n_in; (void)out_size; (void)ws_size;
    const float* x      = (const float*)d_in[0];
    const float* mask   = (const float*)d_in[1];
    const float* gmask  = (const float*)d_in[2];
    const float* qkv_w  = (const float*)d_in[3];
    const float* kvg_w  = (const float*)d_in[4];
    const float* proj_w = (const float*)d_in[5];
    const float* proj_b = (const float*)d_in[6];
    float* out = (float*)d_out;

    char* ws = (char*)d_ws;
    size_t off = 0;
    auto alloc = [&](size_t bytes) {
        void* p = ws + off;
        off = (off + bytes + 255) & ~(size_t)255;
        return p;
    };
    __bf16* xb    = (__bf16*)alloc((size_t)MTOT * CC * 2);       // x bf16; reused as y
    __bf16* qwb   = (__bf16*)alloc((size_t)CC * 3 * CC * 2);     // qkv_w bf16
    __bf16* pwb   = (__bf16*)alloc((size_t)CC * CC * 2);         // proj_w bf16
    __bf16* qkvb  = (__bf16*)alloc((size_t)MTOT * 3 * CC * 2);   // qkv bf16
    float*  x_cls = (float*)alloc((size_t)BQ * CC * 4);
    float*  kvg   = (float*)alloc((size_t)BQ * 2 * CC * 4);
    __bf16* y = xb;  // xb fully consumed by GEMM1 before y is written

    // 1) conversions
    {
        int n = MTOT * CC;
        k_f32_to_bf16<<<(n + 255) / 256, 256, 0, stream>>>(x, xb, n);
        n = CC * 3 * CC;
        k_f32_to_bf16<<<(n + 255) / 256, 256, 0, stream>>>(qkv_w, qwb, n);
        n = CC * CC;
        k_f32_to_bf16<<<(n + 255) / 256, 256, 0, stream>>>(proj_w, pwb, n);
    }
    // 2) qkv = x @ qkv_w  (WMMA bf16)
    {
        dim3 g((MTOT + 127) / 128, (3 * CC) / 64);
        k_gemm_bf16<true, false><<<g, 256, 0, stream>>>(xb, qwb, nullptr, qkvb,
                                                        nullptr, MTOT, 3 * CC, CC);
    }
    // 3) window attention (WMMA)
    {
        dim3 g(64, NHD, BQ);
        k_window_attn<<<g, 128, 0, stream>>>(qkvb, mask, y);
    }
    // 4) global cls attention
    {
        dim3 g(NHD, BQ);
        k_cls_attn<<<g, 256, 0, stream>>>(qkvb, mask, gmask, x_cls);
    }
    // 5) kv_g = x_cls @ kv_global_w
    k_kvg<<<(BQ * 2 * CC + 255) / 256, 256, 0, stream>>>(x_cls, kvg_w, kvg);
    // 6) assemble y
    {
        size_t n = (size_t)MTOT * CC;
        k_build_y<<<(unsigned)((n + 255) / 256), 256, 0, stream>>>(y, x_cls, kvg);
    }
    // 7) out = y @ proj_w + proj_b  (WMMA bf16, f32 out)
    {
        dim3 g((MTOT + 127) / 128, CC / 64);
        k_gemm_bf16<false, true><<<g, 256, 0, stream>>>(y, pwb, out, nullptr,
                                                        proj_b, MTOT, CC, CC);
    }
}
